// MMDiTAttention_20564303413794
// MI455X (gfx1250) — matrix operand order repaired and checked by reading, not compiled, not run on hardware
//
#include <hip/hip_runtime.h>
#include <hip/hip_bf16.h>
#include <math.h>
#include <stdint.h>

typedef __attribute__((ext_vector_type(16))) __bf16 bf16x16;
typedef __attribute__((ext_vector_type(8)))  float  f32x8;

#define NUM_HEADS 12
#define HEAD_DIM  128
#define DIM       1536
#define BATCH     2
#define S_IN      2048
#define S_CTX     256
#define NTOK      (S_IN + S_CTX)   /* 2304 */
#define QKV_N     (3 * DIM)        /* 4608 */

// ---------------------------------------------------------------------------
// Fragment helpers (ISA 7.12.2 VGPR layouts, wave32). All fragment addresses
// are multiples of 8 bf16 (16 B), so load them as uint4 to force b128 ops.
//  A (16x32 bf16): lane<16 -> row=lane, K 0..15 ; lane>=16 -> K 16..31 (contiguous k order used here)
//  B (32x16 bf16): lane<16 -> col=lane, K 0..15 ; lane>=16 -> K 16..31 (same contiguous order as A)
//  C/D (16x16 f32): VGPR r -> M = r + 8*(lane>=16), N = lane&15
// ---------------------------------------------------------------------------
__device__ inline f32x8 zero8() {
    f32x8 z;
#pragma unroll
    for (int i = 0; i < 8; ++i) z[i] = 0.0f;
    return z;
}

__device__ inline bf16x16 fragu(const __bf16* pa, const __bf16* pb) {
    union { uint4 u[2]; bf16x16 f; } x;
    x.u[0] = *reinterpret_cast<const uint4*>(pa);
    x.u[1] = *reinterpret_cast<const uint4*>(pb);
    return x.f;
}

__device__ inline bf16x16 fragc(const __bf16* p) {  // 16 contiguous bf16
    union { uint4 u[2]; bf16x16 f; } x;
    x.u[0] = *reinterpret_cast<const uint4*>(p);
    x.u[1] = *reinterpret_cast<const uint4*>(p + 8);
    return x.f;
}

__device__ inline f32x8 wmma_bf16(bf16x16 a, bf16x16 b, f32x8 c) {
    return __builtin_amdgcn_wmma_f32_16x16x32_bf16(
        /*neg_a=*/false, a, /*neg_b=*/false, b,
        /*c_mod=*/(short)0, c, /*reuse_a=*/false, /*reuse_b=*/false);
}

// ---------------------------------------------------------------------------
// CDNA5 async global->LDS copy (ASYNCcnt-tracked; ISA 10. / 15.18.3 op 98).
// Each lane supplies its own 32-bit LDS dest address and 64-bit global addr.
// LDS address = low 32 bits of the generic pointer (ISA 10.2 aperture rule).
// ---------------------------------------------------------------------------
__device__ inline void async_ld_b128(uint32_t lds_addr, unsigned long long gaddr) {
    asm volatile("global_load_async_to_lds_b128 %0, %1, off"
                 :: "v"(lds_addr), "v"(gaddr) : "memory");
}
__device__ inline void wait_asynccnt0() {
#if __has_builtin(__builtin_amdgcn_s_wait_asynccnt)
    __builtin_amdgcn_s_wait_asynccnt(0);
#else
    asm volatile("s_wait_asynccnt 0x0" ::: "memory");
#endif
}

// ---------------------------------------------------------------------------
// Generic double-buffered bf16-WMMA GEMM body (M-tile 128, N-tile 128, K=DIM).
// Waves: 8 = 4(M) x 2(N); each wave 32x64 -> 2x4 accumulator tiles.
// Staging: register-block global loads (full MLP), then cvt -> LDS; LDS double
// buffered so stage(i+1) global loads overlap WMMA(i). One barrier / K-step.
// ---------------------------------------------------------------------------
#define GEMM_BODY(X, W, NW, Asm, Bsm, acc)                                     \
    float4 ra[4], rb[4];                                                       \
    {   /* prologue: stage chunk 0 */                                          \
        _Pragma("unroll")                                                      \
        for (int qd = 0; qd < 4; ++qd) {                                       \
            int u = tid + 256 * qd;                                            \
            int row = u >> 3, kc = (u & 7) * 4;                                \
            ra[qd] = *reinterpret_cast<const float4*>(                         \
                (X) + (size_t)(m0 + row) * DIM + kc);                          \
            int kr = u >> 5, cc = (u & 31) * 4;                                \
            rb[qd] = *reinterpret_cast<const float4*>(                         \
                (W) + (size_t)kr * (NW) + n0 + cc);                            \
        }                                                                      \
        _Pragma("unroll")                                                      \
        for (int qd = 0; qd < 4; ++qd) {                                       \
            int u = tid + 256 * qd;                                            \
            int row = u >> 3, kc = (u & 7) * 4;                                \
            __bf16* d = &Asm[0][row * 32 + kc];                                \
            d[0] = (__bf16)ra[qd].x; d[1] = (__bf16)ra[qd].y;                  \
            d[2] = (__bf16)ra[qd].z; d[3] = (__bf16)ra[qd].w;                  \
            int kr = u >> 5, cc = (u & 31) * 4;                                \
            Bsm[0][(cc + 0) * 32 + kr] = (__bf16)rb[qd].x;                     \
            Bsm[0][(cc + 1) * 32 + kr] = (__bf16)rb[qd].y;                     \
            Bsm[0][(cc + 2) * 32 + kr] = (__bf16)rb[qd].z;                     \
            Bsm[0][(cc + 3) * 32 + kr] = (__bf16)rb[qd].w;                     \
        }                                                                      \
        __syncthreads();                                                       \
    }                                                                          \
    const int nk = DIM / 32;                                                   \
    for (int it = 0; it < nk; ++it) {                                          \
        const int buf = it & 1;                                                \
        const int kk1 = (it + 1) * 32;                                         \
        if (it + 1 < nk) {  /* issue next-chunk global loads (no wait) */      \
            _Pragma("unroll")                                                  \
            for (int qd = 0; qd < 4; ++qd) {                                   \
                int u = tid + 256 * qd;                                        \
                int row = u >> 3, kc = (u & 7) * 4;                            \
                ra[qd] = *reinterpret_cast<const float4*>(                     \
                    (X) + (size_t)(m0 + row) * DIM + kk1 + kc);                \
                int kr = u >> 5, cc = (u & 31) * 4;                            \
                rb[qd] = *reinterpret_cast<const float4*>(                     \
                    (W) + (size_t)(kk1 + kr) * (NW) + n0 + cc);                \
            }                                                                  \
            if (kk1 + 32 < DIM)                                                \
                __builtin_prefetch((W) + (size_t)(kk1 + 32) * (NW) + n0, 0, 1);\
        }                                                                      \
        _Pragma("unroll")                                                      \
        for (int ms = 0; ms < 2; ++ms) {  /* WMMA on current buffer */         \
            int arow = wm * 32 + ms * 16 + l15;                                \
            bf16x16 af = fragc(&Asm[buf][arow * 32 + half * 16]);              \
            _Pragma("unroll")                                                  \
            for (int t = 0; t < 4; ++t) {                                      \
                int bcol = wn * 64 + t * 16 + l15;                             \
                bf16x16 bf = fragc(&Bsm[buf][bcol * 32 + half * 16]);          \
                acc[ms][t] = wmma_bf16(af, bf, acc[ms][t]);                    \
            }                                                                  \
        }                                                                      \
        if (it + 1 < nk) {  /* commit next chunk into other buffer */          \
            _Pragma("unroll")                                                  \
            for (int qd = 0; qd < 4; ++qd) {                                   \
                int u = tid + 256 * qd;                                        \
                int row = u >> 3, kc = (u & 7) * 4;                            \
                __bf16* d = &Asm[buf ^ 1][row * 32 + kc];                      \
                d[0] = (__bf16)ra[qd].x; d[1] = (__bf16)ra[qd].y;              \
                d[2] = (__bf16)ra[qd].z; d[3] = (__bf16)ra[qd].w;              \
                int kr = u >> 5, cc = (u & 31) * 4;                            \
                Bsm[buf ^ 1][(cc + 0) * 32 + kr] = (__bf16)rb[qd].x;           \
                Bsm[buf ^ 1][(cc + 1) * 32 + kr] = (__bf16)rb[qd].y;           \
                Bsm[buf ^ 1][(cc + 2) * 32 + kr] = (__bf16)rb[qd].z;           \
                Bsm[buf ^ 1][(cc + 3) * 32 + kr] = (__bf16)rb[qd].w;           \
            }                                                                  \
        }                                                                      \
        __syncthreads();                                                       \
    }

// ---------------------------------------------------------------------------
// Kernel 1: QKV projection  C = X @ W + b  (K = DIM, N = 3*DIM)
// q,k -> f32 buffers (pre-norm precision); v -> bf16, transposed per (b,h)
// to [d][token] so attention can async-copy contiguous V rows.
// ---------------------------------------------------------------------------
__global__ __launch_bounds__(256)
void qkv_gemm_kernel(const float* __restrict__ X, const float* __restrict__ W,
                     const float* __restrict__ bias,
                     float* __restrict__ qf, float* __restrict__ kf,
                     __bf16* __restrict__ vtb, int s_len, int tok_off) {
    __shared__ __bf16 Asm[2][128 * 32];
    __shared__ __bf16 Bsm[2][128 * 32];
    const int tid = threadIdx.x;
    const int lane = tid & 31, w = tid >> 5;
    const int wm = w >> 1, wn = w & 1;
    const int half = lane >> 4, l15 = lane & 15;
    const int m0 = blockIdx.x * 128;
    const int n0 = blockIdx.y * 128;

    f32x8 acc[2][4];
#pragma unroll
    for (int i = 0; i < 2; ++i)
#pragma unroll
        for (int j = 0; j < 4; ++j) acc[i][j] = zero8();

    GEMM_BODY(X, W, QKV_N, Asm, Bsm, acc)

    // epilogue: bias add + scatter into q/k/v at concatenated token position
#pragma unroll
    for (int ms = 0; ms < 2; ++ms) {
#pragma unroll
        for (int t = 0; t < 4; ++t) {
            int col = n0 + wn * 64 + t * 16 + l15;
            int which = col / DIM;                  // uniform within 128-tile
            int c = col - which * DIM;
            float bv = bias[col];
            if (which == 2) {                       // V: bf16, [b][h][d][tok]
                int hh = c >> 7, d = c & 127;
#pragma unroll
                for (int r = 0; r < 8; ++r) {
                    int gm = m0 + wm * 32 + ms * 16 + r + 8 * half;
                    int bidx = gm / s_len;
                    int s = gm - bidx * s_len;
                    vtb[(((size_t)bidx * NUM_HEADS + hh) * HEAD_DIM + d) * NTOK
                        + tok_off + s] = (__bf16)(acc[ms][t][r] + bv);
                }
            } else {                                // Q/K: f32
                float* dst = which ? kf : qf;
#pragma unroll
                for (int r = 0; r < 8; ++r) {
                    int gm = m0 + wm * 32 + ms * 16 + r + 8 * half;
                    int bidx = gm / s_len;
                    int s = gm - bidx * s_len;
                    size_t drow = (size_t)(bidx * NTOK + tok_off + s);
                    dst[drow * DIM + c] = acc[ms][t][r] + bv;
                }
            }
        }
    }
}

// ---------------------------------------------------------------------------
// Kernel 2: fused RMS-norm + RoPE; reads f32, writes roped bf16.
// ---------------------------------------------------------------------------
__global__ __launch_bounds__(256)
void rmsnorm_rope_kernel(const float* __restrict__ xin,
                         __bf16* __restrict__ xout,
                         const float* __restrict__ sc_in,
                         const float* __restrict__ sc_ctx) {
    __shared__ float red[8];
    __shared__ float row[DIM];
    const int g = blockIdx.x;                 // 0 .. BATCH*NTOK-1
    const int b = g / NTOK, t = g - b * NTOK;
    const float* scale = (t < S_CTX) ? sc_ctx : sc_in;
    const float* xr = xin + (size_t)g * DIM;
    __bf16* xo = xout + (size_t)g * DIM;
    const int tid = threadIdx.x;

    float ss = 0.0f;
#pragma unroll
    for (int i = 0; i < 6; ++i) {
        int c = tid + 256 * i;
        float vv = xr[c];
        ss += vv * vv;
        row[c] = vv;
    }
#pragma unroll
    for (int msk = 1; msk < 32; msk <<= 1) ss += __shfl_xor(ss, msk, 32);
    if ((tid & 31) == 0) red[tid >> 5] = ss;
    __syncthreads();
    float tot = 0.0f;
#pragma unroll
    for (int i = 0; i < 8; ++i) tot += red[i];
    float rs = __frsqrt_rn(tot / (float)DIM + 1e-6f);
#pragma unroll
    for (int i = 0; i < 6; ++i) {
        int c = tid + 256 * i;
        row[c] = row[c] * rs * scale[c];
    }
    __syncthreads();
    // RoPE: 768 (x1,x2) pairs; inv_freq = 10000^{-j/64}
#pragma unroll
    for (int i = 0; i < 3; ++i) {
        int p = tid + 256 * i;
        int hh = p >> 6, j = p & 63;
        int i1 = hh * HEAD_DIM + j, i2 = i1 + 64;
        float inv = __expf(-(float)j * (9.210340371976184f / 64.0f));
        float ang = (float)t * inv;
        float s, c;
        __sincosf(ang, &s, &c);
        float x1 = row[i1], x2 = row[i2];
        xo[i1] = (__bf16)(x1 * c - x2 * s);
        xo[i2] = (__bf16)(x2 * c + x1 * s);
    }
}

// ---------------------------------------------------------------------------
// Kernel 3: flash attention on bf16 q/k/vt. grid = (NTOK/128, B*H).
// K and Vt tiles staged via ASYNC global->LDS DMA, double-buffered.
// Per-lane DMA addresses are maintained in registers (strength-reduced).
// ---------------------------------------------------------------------------
__global__ __launch_bounds__(256)
void attention_kernel(const __bf16* __restrict__ qb,
                      const __bf16* __restrict__ kbf,
                      const __bf16* __restrict__ vtb,
                      float* __restrict__ o) {
    __shared__ __bf16 Ksm[2][32 * 128];     // [key][d]
    __shared__ __bf16 Vsm[2][128 * 32];     // [d][key]  (vt is pre-transposed)
    __shared__ __bf16 Psm[8][16 * 32];      // per-wave P tile [q][key]
    const int tid = threadIdx.x, lane = tid & 31, w = tid >> 5;
    const int half = lane >> 4, l15 = lane & 15;
    const int bh = blockIdx.y, b = bh / NUM_HEADS, h = bh - b * NUM_HEADS;
    const int q0 = blockIdx.x * 128 + w * 16;
    const size_t baseq = (size_t)(b * NTOK) * DIM + h * HEAD_DIM;
    const __bf16* kg = kbf + baseq;                                  // +key*DIM+d
    const __bf16* vg = vtb + (size_t)(b * NUM_HEADS + h) * HEAD_DIM * NTOK; // +d*NTOK+tok

    // Q fragments straight from bf16 global (no conversion in hot loop)
    bf16x16 qf[4];
#pragma unroll
    for (int c = 0; c < 4; ++c) {
        const __bf16* pr = qb + baseq + (size_t)(q0 + l15) * DIM + c * 32 + half * 16;
        qf[c] = fragc(pr);
    }
    f32x8 oacc[8];
#pragma unroll
    for (int t = 0; t < 8; ++t) oacc[t] = zero8();
    float m[8], l[8];
#pragma unroll
    for (int r = 0; r < 8; ++r) { m[r] = -1e30f; l[r] = 0.0f; }
    const float sc = 0.08838834764831845f;  // 1/sqrt(128)

    // ---- per-lane async DMA addresses, strength-reduced across key blocks --
    // K tile:  32 keys x 256B; lane covers (key=tid>>4, seg) and key+16
    // Vt tile: 128 d-rows x 64B; lane covers (d=tid>>2, seg2) and d+64
    const int key = tid >> 4, seg = (tid & 15) * 8;
    const int dd = tid >> 2, sg2 = (tid & 3) * 8;
    unsigned long long gK =
        (unsigned long long)(uintptr_t)(kg + (size_t)key * DIM + seg);
    unsigned long long gV =
        (unsigned long long)(uintptr_t)(vg + (size_t)dd * NTOK + sg2);
    const uint32_t lK = (uint32_t)(uintptr_t)&Ksm[0][key * 128 + seg];
    const uint32_t lV = (uint32_t)(uintptr_t)&Vsm[0][dd * 32 + sg2];
    const unsigned long long K_STEP = (unsigned long long)32 * DIM * 2;  // next block
    const unsigned long long V_STEP = 32 * 2;
    const unsigned long long K_QD1 = (unsigned long long)16 * DIM * 2;   // +16 keys
    const unsigned long long V_QD1 = (unsigned long long)64 * NTOK * 2;  // +64 d rows
    const uint32_t LBUF = 32 * 128 * 2;  // 8192B: second LDS buffer offset
    const uint32_t LQD1 = 4096;          // +16 keys (K) / +64 rows (Vt) in LDS

    auto issue_async = [&](int bf) {
        uint32_t lo = bf ? LBUF : 0u;
        async_ld_b128(lK + lo, gK);
        async_ld_b128(lK + lo + LQD1, gK + K_QD1);
        async_ld_b128(lV + lo, gV);
        async_ld_b128(lV + lo + LQD1, gV + V_QD1);
        gK += K_STEP;
        gV += V_STEP;
    };

    issue_async(0);
    wait_asynccnt0();
    __syncthreads();

    const int nkb = NTOK / 32;
    for (int it = 0; it < nkb; ++it) {
        const int buf = it & 1;
        if (it + 1 < nkb) issue_async(buf ^ 1);

        // S = Q Kt : two 16x16 tiles (keys 0..15, 16..31), 4 K-dim steps
        f32x8 s0 = zero8(), s1 = zero8();
#pragma unroll
        for (int c = 0; c < 4; ++c) {
            bf16x16 k0f = fragc(&Ksm[buf][l15 * 128 + c * 32 + half * 16]);
            bf16x16 k1f = fragc(&Ksm[buf][(16 + l15) * 128 + c * 32 + half * 16]);
            s0 = wmma_bf16(qf[c], k0f, s0);
            s1 = wmma_bf16(qf[c], k1f, s1);
        }

        // online softmax per row (half-wave shuffle reductions)
#pragma unroll
        for (int r = 0; r < 8; ++r) {
            float a0 = s0[r] * sc, a1 = s1[r] * sc;
            float mx = fmaxf(a0, a1);
#pragma unroll
            for (int msk = 1; msk < 16; msk <<= 1)
                mx = fmaxf(mx, __shfl_xor(mx, msk, 32));
            float mnew = fmaxf(m[r], mx);
            float alpha = __expf(m[r] - mnew);
            float p0 = __expf(a0 - mnew), p1 = __expf(a1 - mnew);
            float rsum = p0 + p1;
#pragma unroll
            for (int msk = 1; msk < 16; msk <<= 1)
                rsum += __shfl_xor(rsum, msk, 32);
            l[r] = l[r] * alpha + rsum;
            m[r] = mnew;
#pragma unroll
            for (int t = 0; t < 8; ++t) oacc[t][r] *= alpha;
            Psm[w][(r + 8 * half) * 32 + l15]      = (__bf16)p0;
            Psm[w][(r + 8 * half) * 32 + 16 + l15] = (__bf16)p1;
        }

        // O += P @ V  (same-wave LDS restage: DS ops are in-order per wave)
        bf16x16 pf = fragc(&Psm[w][l15 * 32 + half * 16]);
#pragma unroll
        for (int t = 0; t < 8; ++t) {
            bf16x16 vf = fragc(&Vsm[buf][(t * 16 + l15) * 32 + half * 16]);
            oacc[t] = wmma_bf16(pf, vf, oacc[t]);
        }

        if (it + 1 < nkb) wait_asynccnt0();
        __syncthreads();
    }

    // normalize by l and write out
#pragma unroll
    for (int t = 0; t < 8; ++t) {
#pragma unroll
        for (int r = 0; r < 8; ++r) {
            int row = q0 + r + 8 * half;
            o[baseq + (size_t)row * DIM + t * 16 + l15] = oacc[t][r] / l[r];
        }
    }
}

// ---------------------------------------------------------------------------
// Kernel 4: output projection; W/bias selected per ctx/input row tile.
// ---------------------------------------------------------------------------
__global__ __launch_bounds__(256)
void out_gemm_kernel(const float* __restrict__ A,
                     const float* __restrict__ Wi, const float* __restrict__ bi,
                     const float* __restrict__ Wc, const float* __restrict__ bc,
                     float* __restrict__ oi, float* __restrict__ oc) {
    __shared__ __bf16 Asm[2][128 * 32];
    __shared__ __bf16 Bsm[2][128 * 32];
    const int tid = threadIdx.x;
    const int lane = tid & 31, w = tid >> 5;
    const int wm = w >> 1, wn = w & 1;
    const int half = lane >> 4, l15 = lane & 15;
    const int m0 = blockIdx.x * 128;
    const int n0 = blockIdx.y * 128;
    const bool is_ctx = (m0 % NTOK) < S_CTX;    // uniform per 128-row tile
    const float* W = is_ctx ? Wc : Wi;
    const float* bias = is_ctx ? bc : bi;

    f32x8 acc[2][4];
#pragma unroll
    for (int i = 0; i < 2; ++i)
#pragma unroll
        for (int j = 0; j < 4; ++j) acc[i][j] = zero8();

    GEMM_BODY(A, W, DIM, Asm, Bsm, acc)

#pragma unroll
    for (int ms = 0; ms < 2; ++ms) {
#pragma unroll
        for (int t = 0; t < 4; ++t) {
            int col = n0 + wn * 64 + t * 16 + l15;
            float bv = bias[col];
#pragma unroll
            for (int r = 0; r < 8; ++r) {
                int gm = m0 + wm * 32 + ms * 16 + r + 8 * half;
                int bb = gm / NTOK;
                int tt = gm - bb * NTOK;
                float val = acc[ms][t][r] + bv;
                if (is_ctx)
                    oc[((size_t)bb * S_CTX + tt) * DIM + col] = val;
                else
                    oi[((size_t)bb * S_IN + (tt - S_CTX)) * DIM + col] = val;
            }
        }
    }
}

// ---------------------------------------------------------------------------
extern "C" void kernel_launch(void* const* d_in, const int* in_sizes, int n_in,
                              void* d_out, int out_size, void* d_ws,
                              size_t ws_size, hipStream_t stream) {
    const float* input     = (const float*)d_in[0];
    const float* context   = (const float*)d_in[1];
    const float* W_qkv_in  = (const float*)d_in[2];
    const float* b_qkv_in  = (const float*)d_in[3];
    const float* W_qkv_ctx = (const float*)d_in[4];
    const float* b_qkv_ctx = (const float*)d_in[5];
    const float* qs_in     = (const float*)d_in[6];
    const float* ks_in     = (const float*)d_in[7];
    const float* qs_ctx    = (const float*)d_in[8];
    const float* ks_ctx    = (const float*)d_in[9];
    const float* W_out_in  = (const float*)d_in[10];
    const float* b_out_in  = (const float*)d_in[11];
    const float* W_out_ctx = (const float*)d_in[12];
    const float* b_out_ctx = (const float*)d_in[13];

    const size_t NTOT = (size_t)BATCH * NTOK * DIM;  // 7,077,888 elements
    char* base = (char*)d_ws;
    float*  qf32 = (float*)base;                     // f32 q (pre-norm)
    float*  kf32 = (float*)(base + NTOT * 4);        // f32 k (pre-norm)
    __bf16* qb   = (__bf16*)(base + 2 * NTOT * 4);   // roped bf16 q
    __bf16* kb   = qb + NTOT;                        // roped bf16 k
    __bf16* vt   = kb + NTOT;                        // bf16 v, [b][h][d][tok]
    float*  ao   = qf32;                             // attn out aliases dead q

    float* out_in  = (float*)d_out;                       // (B, S_IN, DIM)
    float* out_ctx = out_in + (size_t)BATCH * S_IN * DIM; // (B, S_CTX, DIM)

    // 1) QKV projections (input tokens land at offset S_CTX, ctx at 0)
    qkv_gemm_kernel<<<dim3((BATCH * S_IN) / 128, QKV_N / 128), 256, 0, stream>>>(
        input, W_qkv_in, b_qkv_in, qf32, kf32, vt, S_IN, S_CTX);
    qkv_gemm_kernel<<<dim3((BATCH * S_CTX) / 128, QKV_N / 128), 256, 0, stream>>>(
        context, W_qkv_ctx, b_qkv_ctx, qf32, kf32, vt, S_CTX, 0);

    // 2) RMS-norm + RoPE; f32 -> roped bf16
    rmsnorm_rope_kernel<<<BATCH * NTOK, 256, 0, stream>>>(qf32, qb, qs_in, qs_ctx);
    rmsnorm_rope_kernel<<<BATCH * NTOK, 256, 0, stream>>>(kf32, kb, ks_in, ks_ctx);

    // 3) attention (async LDS staging of K and pre-transposed V)
    attention_kernel<<<dim3(NTOK / 128, BATCH * NUM_HEADS), 256, 0, stream>>>(
        qb, kb, vt, ao);

    // 4) output projections
    out_gemm_kernel<<<dim3((BATCH * NTOK) / 128, DIM / 128), 256, 0, stream>>>(
        ao, W_out_in, b_out_in, W_out_ctx, b_out_ctx, out_in, out_ctx);
}
